// StackEncoder_64828236366680
// MI455X (gfx1250) — matrix operand order repaired and checked by reading, not compiled, not run on hardware
//
#include <hip/hip_runtime.h>

// ---------------------------------------------------------------------------
// SPINN (stack encoder) for MI455X / gfx1250, wave32 + WMMA bf16.
// Heavy math per step is folded into two GEMMs driven by
// v_wmma_f32_16x16x32_bf16 (fp32 accumulate), weights pre-packed once per
// launch into bf16 K-major layout. GEMM tiles are staged into LDS with
// GLOBAL_LOAD_ASYNC_TO_LDS_B128 (ASYNCcnt) instead of load+ds_store.
// Entire working set (~40MB) sits in the 192MB L2; the serial 63-step chain
// is latency-bound, so per-step kernels are kept minimal.
// ---------------------------------------------------------------------------

typedef __bf16 bf16;
typedef bf16  v8bf  __attribute__((ext_vector_type(8)));
typedef bf16  v16bf __attribute__((ext_vector_type(16)));
typedef float v8f   __attribute__((ext_vector_type(8)));

#define B_    128
#define L_    32
#define H_    512
#define XD_   64
#define T_    63
#define KTREE 1088              // 64 (x) + 512 (h_top) + 512 (h_second)
#define NTREE 2048              // 4 gates x 512
#define KTRK  1600              // 1536 (tin) + 64 (th)
#define NTRK  256               // 4 gates x 64
#define LDST  40                // padded LDS row stride in halves (80B, 16B aligned)

__device__ __forceinline__ float sigf(float x) { return 1.0f / (1.0f + __expf(-x)); }

// Low 32 bits of a generic LDS pointer == DS-relative address (ISA 10.2:
// for the LDS aperture).
__device__ __forceinline__ unsigned lds_off(const void* p) {
    return (unsigned)(unsigned long long)p;
}

// ---------------------------------------------------------------------------
// Generic bf16 WMMA GEMM:  C[M x N] = A[M x K] * B[N x K]^T   (B is K-major
// per row n, i.e. row-major N x K). Block = 256 threads = 8 waves, tile 64x64.
// Each wave owns one 16-row strip and two 16-col tiles -> 2 WMMA per K-chunk.
// Tiles are staged global->LDS with the gfx1250 async DMA path (ASYNCcnt).
// ---------------------------------------------------------------------------
__global__ __launch_bounds__(256)
void gemm_bf16_wmma(const bf16* __restrict__ A, const bf16* __restrict__ Bm,
                    float* __restrict__ C, int M, int N, int K) {
    __shared__ bf16 lA[64 * LDST];
    __shared__ bf16 lB[64 * LDST];
    const int tid  = threadIdx.x;
    const int gm0  = blockIdx.y * 64;
    const int gn0  = blockIdx.x * 64;
    const int w    = tid >> 5;
    const int lane = tid & 31;
    const int half = lane >> 4;       // 0: lanes 0-15, 1: lanes 16-31
    const int l15  = lane & 15;
    const int mt   = (w & 3) * 16;    // wave's M sub-tile in block tile
    const int nt   = (w >> 2) * 32;   // wave's N sub-tiles (nt, nt+16)
    const int rowL = tid >> 2;        // 64 rows, 4 threads/row
    const int ch   = (tid & 3) * 8;   // 8-half chunk within 32-wide K slab

    const unsigned la = lds_off(lA + rowL * LDST + ch);
    const unsigned lb = lds_off(lB + rowL * LDST + ch);

    v8f acc0 = {}; v8f acc1 = {};
    const int KC = K >> 5;            // K is a multiple of 32 (1088, 1600)
    for (int kc = 0; kc < KC; ++kc) {
        const int kb = kc * 32 + ch;
        const bf16* gA = A  + (size_t)(gm0 + rowL) * K + kb;
        const bf16* gB = Bm + (size_t)(gn0 + rowL) * K + kb;
        // async DMA: 16B per lane, global -> LDS, tracked by ASYNCcnt
        asm volatile("global_load_async_to_lds_b128 %0, %1, off"
                     :: "v"(la), "v"(gA) : "memory");
        asm volatile("global_load_async_to_lds_b128 %0, %1, off"
                     :: "v"(lb), "v"(gB) : "memory");
        if (kc + 1 < KC) {   // global_prefetch_b8 for the next K slab
            __builtin_prefetch(gA + 32, 0, 0);
            __builtin_prefetch(gB + 32, 0, 0);
        }
        asm volatile("s_wait_asynccnt 0x0" ::: "memory");
        __syncthreads();

        // A fragment (16x32 bf16): lanes 0-15 -> K 0..15; 16-31 -> K 16..31
        const v8bf* ra = (const v8bf*)(lA + (mt + l15) * LDST);
        v8bf a0 = ra[half * 2], a1 = ra[half * 2 + 1];
        v16bf af = __builtin_shufflevector(a0, a1, 0,1,2,3,4,5,6,7,8,9,10,11,12,13,14,15);

        // B fragments (32x16 bf16): lanes 0-15 col n hold K 0..15; lanes 16-31 K 16..31
        const v8bf* rb0 = (const v8bf*)(lB + (nt + l15) * LDST);
        v8bf b00 = rb0[half * 2], b01 = rb0[half * 2 + 1];
        v16bf bf0 = __builtin_shufflevector(b00, b01, 0,1,2,3,4,5,6,7,8,9,10,11,12,13,14,15);
        const v8bf* rb1 = (const v8bf*)(lB + (nt + 16 + l15) * LDST);
        v8bf b10 = rb1[half * 2], b11 = rb1[half * 2 + 1];
        v16bf bf1 = __builtin_shufflevector(b10, b11, 0,1,2,3,4,5,6,7,8,9,10,11,12,13,14,15);

        acc0 = __builtin_amdgcn_wmma_f32_16x16x32_bf16(false, af, false, bf0,
                                                       (short)0, acc0, false, false);
        acc1 = __builtin_amdgcn_wmma_f32_16x16x32_bf16(false, af, false, bf1,
                                                       (short)0, acc1, false, false);
        __syncthreads();
    }
    // C/D layout: VGPR r -> (M = r + 8*half, N = l15)
    const int mrow = gm0 + mt + half * 8;
    const int nc   = gn0 + nt + l15;
    for (int r = 0; r < 8; ++r) {
        C[(size_t)(mrow + r) * N + nc]      = acc0[r];
        C[(size_t)(mrow + r) * N + nc + 16] = acc1[r];
    }
}

// ---------------------------------------------------------------------------
// One-time weight packing (bf16, K-major rows). Tree gate order along N:
// [0,512)=i, [512,1024)=o, [1024,1536)=f (uses W_o: reference bug kept),
// [1536,2048)=u.  K: [0,64)=x, [64,576)=h_top (U_*l), [576,1088)=h_2nd (U_*r).
// ---------------------------------------------------------------------------
__global__ void prep_tree_w(const float* __restrict__ Wi, const float* __restrict__ Wo,
                            const float* __restrict__ Wu,
                            const float* __restrict__ Uil, const float* __restrict__ Uir,
                            const float* __restrict__ Ufl, const float* __restrict__ Ufr,
                            const float* __restrict__ Uol, const float* __restrict__ Uor,
                            const float* __restrict__ Uul, const float* __restrict__ Uur,
                            bf16* __restrict__ WB) {
    int idx = blockIdx.x * blockDim.x + threadIdx.x;
    if (idx >= NTREE * KTREE) return;
    int n = idx / KTREE, k = idx % KTREE;
    int g = n >> 9, r = n & 511;
    float v;
    if (k < 64) {
        const float* W = (g == 0) ? Wi : (g == 3) ? Wu : Wo;   // o and f both use W_o
        v = W[r * 64 + k];
    } else if (k < 576) {
        const float* U = (g == 0) ? Uil : (g == 1) ? Uol : (g == 2) ? Ufl : Uul;
        v = U[r * 512 + (k - 64)];
    } else {
        const float* U = (g == 0) ? Uir : (g == 1) ? Uor : (g == 2) ? Ufr : Uur;
        v = U[r * 512 + (k - 576)];
    }
    WB[(size_t)n * KTREE + k] = (bf16)v;
}

__global__ void prep_track_w(const float* __restrict__ tWih, const float* __restrict__ tWhh,
                             bf16* __restrict__ TB) {
    int idx = blockIdx.x * blockDim.x + threadIdx.x;
    if (idx >= NTRK * KTRK) return;
    int n = idx / KTRK, k = idx % KTRK;
    float v = (k < 1536) ? tWih[n * 1536 + k] : tWhh[n * 64 + (k - 1536)];
    TB[(size_t)n * KTRK + k] = (bf16)v;
}

// zero stack slot 0, seed tracking state, zero int state
__global__ void init_state(const float* __restrict__ th0, const float* __restrict__ tc0,
                           float* __restrict__ stack, float* __restrict__ th,
                           float* __restrict__ tc, int* __restrict__ qlen,
                           int* __restrict__ bp) {
    int idx = blockIdx.x * blockDim.x + threadIdx.x;
    if (idx < 128 * 1024) {
        int b = idx >> 10, k = idx & 1023;
        stack[((size_t)b * 64 + 0) * 1024 + k] = 0.f;
    } else if (idx < 128 * 1024 + 8192) {
        int j = idx - 131072; th[j] = th0[j];
    } else if (idx < 128 * 1024 + 16384) {
        int j = idx - 139264; tc[j] = tc0[j];
    } else if (idx < 128 * 1024 + 16384 + 128) {
        qlen[idx - 147456] = 0;
    } else if (idx < 128 * 1024 + 16384 + 256) {
        bp[idx - 147584] = 0;
    }
}

// S1: gather stack/buffer rows, convert to bf16 GEMM inputs Zt (tracking) and
// Z cols [64,1088) (tree). Writes per-row ctrl = {sp_top, sp_2nd, bufslot}.
__global__ void build_inputs(const float* __restrict__ seq, const float* __restrict__ stack,
                             const float* __restrict__ thc, const int* __restrict__ queue,
                             const int* __restrict__ qlen, const int* __restrict__ bp,
                             bf16* __restrict__ Zt, bf16* __restrict__ Z,
                             int* __restrict__ ctrl) {
    int idx = blockIdx.x * blockDim.x + threadIdx.x;
    if (idx >= 128 * 2624) return;
    int b = idx / 2624, col = idx % 2624;
    int ql  = qlen[b];
    int sp1 = (ql >= 1) ? queue[b * 64 + ql - 1] : 0;   // top of stack
    int sp2 = (ql >= 2) ? queue[b * 64 + ql - 2] : 0;   // second
    int bs  = (bp[b] < L_) ? bp[b] : -1;                // -1 -> zero pad slot
    if (col == 0) { ctrl[b * 4] = sp1; ctrl[b * 4 + 1] = sp2; ctrl[b * 4 + 2] = bs; }
    const float* st = stack + (size_t)b * 64 * 1024;
    if (col < 1600) {
        float v;
        if (col < 512)       v = (bs >= 0) ? seq[((size_t)b * L_ + bs) * 1024 + col] : 0.f;
        else if (col < 1024) v = st[sp1 * 1024 + (col - 512)];
        else if (col < 1536) v = st[sp2 * 1024 + (col - 1024)];
        else                 v = thc[b * 64 + (col - 1536)];
        Zt[(size_t)b * KTRK + col] = (bf16)v;
    } else {
        int j = col - 1600;                              // 0..1023
        float v = (j < 512) ? st[sp1 * 1024 + j] : st[sp2 * 1024 + (j - 512)];
        Z[(size_t)b * KTREE + 64 + j] = (bf16)v;
    }
}

// S3: tracking LSTMCell elementwise; writes th/tc (ping-pong) and Z cols [0,64)
__global__ void track_cell(const float* __restrict__ GT, const float* __restrict__ tbih,
                           const float* __restrict__ tbhh, const float* __restrict__ tcc,
                           float* __restrict__ thn, float* __restrict__ tcn,
                           bf16* __restrict__ Z) {
    int idx = blockIdx.x * blockDim.x + threadIdx.x;
    if (idx >= 128 * 64) return;
    int b = idx >> 6, j = idx & 63;
    const float* g = GT + b * 256;
    float gi = g[j]       + tbih[j]       + tbhh[j];
    float gf = g[64 + j]  + tbih[64 + j]  + tbhh[64 + j];
    float gg = g[128 + j] + tbih[128 + j] + tbhh[128 + j];
    float go = g[192 + j] + tbih[192 + j] + tbhh[192 + j];
    float c2 = sigf(gf) * tcc[b * 64 + j] + sigf(gi) * tanhf(gg);
    float h2 = sigf(go) * tanhf(c2);
    tcn[b * 64 + j] = c2;
    thn[b * 64 + j] = h2;
    Z[(size_t)b * KTREE + j] = (bf16)h2;
}

// S5: tree cell elementwise + stack push (shift copies buffer row). At t==T
// also emits the final hidden state to d_out.
__global__ void tree_cell_k(const float* __restrict__ G, const float* __restrict__ bi,
                            const float* __restrict__ bff, const float* __restrict__ bo,
                            const float* __restrict__ bu, const float* __restrict__ seq,
                            const int* __restrict__ trans, const int* __restrict__ ctrl,
                            float* __restrict__ stack, float* __restrict__ out, int t) {
    int idx = blockIdx.x * blockDim.x + threadIdx.x;
    if (idx >= 128 * 512) return;
    int b = idx >> 9, n = idx & 511;
    int sp1 = ctrl[b * 4], sp2 = ctrl[b * 4 + 1], bs = ctrl[b * 4 + 2];
    const float* Gb = G + (size_t)b * 2048;
    float i = sigf(Gb[n] + bi[n]);
    float o = sigf(Gb[512 + n] + bo[n]);
    float f = sigf(Gb[1024 + n] + bff[n]);
    float u = tanhf(Gb[1536 + n] + bu[n]);
    float* st = stack + (size_t)b * 64 * 1024;
    float cl = st[sp1 * 1024 + 512 + n];
    float cr = st[sp2 * 1024 + 512 + n];
    float c  = i * u + f * (cl + cr);     // shared forget gate (reference bug kept)
    float h  = o * tanhf(c);
    int red = trans[b * T_ + (t - 1)];
    float nh, nc;
    if (red == 1) { nh = h; nc = c; }
    else {
        nh = (bs >= 0) ? seq[((size_t)b * L_ + bs) * 1024 + n]       : 0.f;
        nc = (bs >= 0) ? seq[((size_t)b * L_ + bs) * 1024 + 512 + n] : 0.f;
    }
    st[t * 1024 + n]       = nh;
    st[t * 1024 + 512 + n] = nc;
    if (t == T_) out[b * 512 + n] = nh;
}

// S6: per-row integer parser state (runs after all readers of old state)
__global__ void update_state(const int* __restrict__ trans, int* __restrict__ queue,
                             int* __restrict__ qlen, int* __restrict__ bp, int t) {
    int b = blockIdx.x * blockDim.x + threadIdx.x;
    if (b >= 128) return;
    int red = (trans[b * T_ + (t - 1)] == 1);
    int ql = qlen[b];
    int q2 = red ? ql - 2 : ql;
    if (q2 < 0) q2 = 0;
    queue[b * 64 + q2] = t;
    qlen[b] = q2 + 1;
    bp[b] += red ? 0 : 1;
}

// ---------------------------------------------------------------------------
extern "C" void kernel_launch(void* const* d_in, const int* in_sizes, int n_in,
                              void* d_out, int out_size, void* d_ws, size_t ws_size,
                              hipStream_t stream) {
    (void)in_sizes; (void)n_in; (void)out_size; (void)ws_size;
    const float* seq   = (const float*)d_in[0];
    const int*   trans = (const int*)  d_in[1];
    const float* Wi  = (const float*)d_in[2];
    /* d_in[3] = W_f is unused: reference forget gate uses W_o */
    const float* Wo  = (const float*)d_in[4];
    const float* Wu  = (const float*)d_in[5];
    const float* Uil = (const float*)d_in[6],  *Uir = (const float*)d_in[7];
    const float* Ufl = (const float*)d_in[8],  *Ufr = (const float*)d_in[9];
    const float* Uol = (const float*)d_in[10], *Uor = (const float*)d_in[11];
    const float* Uul = (const float*)d_in[12], *Uur = (const float*)d_in[13];
    const float* bi  = (const float*)d_in[14], *bff = (const float*)d_in[15];
    const float* bo  = (const float*)d_in[16], *bu  = (const float*)d_in[17];
    const float* tWih= (const float*)d_in[18], *tWhh= (const float*)d_in[19];
    const float* tbih= (const float*)d_in[20], *tbhh= (const float*)d_in[21];
    const float* th0 = (const float*)d_in[22], *tc0 = (const float*)d_in[23];

    char* ws = (char*)d_ws;
    size_t off = 0;
    auto alloc = [&](size_t bytes) {
        void* p = ws + off;
        off += (bytes + 255) & ~(size_t)255;
        return p;
    };
    float* stack = (float*)alloc((size_t)B_ * 64 * 1024 * 4);   // 33.5 MB
    bf16*  WB    = (bf16*) alloc((size_t)NTREE * KTREE * 2);    // 4.45 MB
    bf16*  TB    = (bf16*) alloc((size_t)NTRK * KTRK * 2);      // 0.82 MB
    bf16*  Z     = (bf16*) alloc((size_t)B_ * KTREE * 2);
    bf16*  Zt    = (bf16*) alloc((size_t)B_ * KTRK * 2);
    float* G     = (float*)alloc((size_t)B_ * NTREE * 4);
    float* GT    = (float*)alloc((size_t)B_ * NTRK * 4);
    float* thb0  = (float*)alloc((size_t)B_ * XD_ * 4);
    float* thb1  = (float*)alloc((size_t)B_ * XD_ * 4);
    float* tcb0  = (float*)alloc((size_t)B_ * XD_ * 4);
    float* tcb1  = (float*)alloc((size_t)B_ * XD_ * 4);
    int*   queue = (int*)  alloc((size_t)B_ * 64 * 4);
    int*   qlen  = (int*)  alloc((size_t)B_ * 4);
    int*   bp    = (int*)  alloc((size_t)B_ * 4);
    int*   ctrl  = (int*)  alloc((size_t)B_ * 4 * 4);
    float* thbuf[2] = { thb0, thb1 };
    float* tcbuf[2] = { tcb0, tcb1 };

    // one-time packing + state init
    prep_tree_w<<<(NTREE * KTREE + 255) / 256, 256, 0, stream>>>(
        Wi, Wo, Wu, Uil, Uir, Ufl, Ufr, Uol, Uor, Uul, Uur, WB);
    prep_track_w<<<(NTRK * KTRK + 255) / 256, 256, 0, stream>>>(tWih, tWhh, TB);
    init_state<<<(147712 + 255) / 256, 256, 0, stream>>>(th0, tc0, stack,
                                                         thbuf[0], tcbuf[0], qlen, bp);

    for (int t = 1; t <= T_; ++t) {
        const float* thc = thbuf[(t - 1) & 1];
        const float* tcc = tcbuf[(t - 1) & 1];
        float* thn = thbuf[t & 1];
        float* tcn = tcbuf[t & 1];

        build_inputs<<<(B_ * 2624 + 255) / 256, 256, 0, stream>>>(
            seq, stack, thc, queue, qlen, bp, Zt, Z, ctrl);
        gemm_bf16_wmma<<<dim3(NTRK / 64, B_ / 64), 256, 0, stream>>>(
            Zt, TB, GT, B_, NTRK, KTRK);
        track_cell<<<(B_ * XD_ + 255) / 256, 256, 0, stream>>>(
            GT, tbih, tbhh, tcc, thn, tcn, Z);
        gemm_bf16_wmma<<<dim3(NTREE / 64, B_ / 64), 256, 0, stream>>>(
            Z, WB, G, B_, NTREE, KTREE);
        tree_cell_k<<<(B_ * H_ + 255) / 256, 256, 0, stream>>>(
            G, bi, bff, bo, bu, seq, trans, ctrl, stack, (float*)d_out, t);
        update_state<<<1, 128, 0, stream>>>(trans, queue, qlen, bp, t);
    }
}
